// ALiBiAttention_83683142795437
// MI455X (gfx1250) — matrix operand order repaired and checked by reading, not compiled, not run on hardware
//
#include <hip/hip_runtime.h>
#include <hip/hip_bf16.h>

// ---------------------------------------------------------------------------
// ALiBi causal attention for MI455X (gfx1250, wave32, WMMA 16x16x32 f16).
//   f32->f16 convert -> QKV projections (WMMA GEMM, async-LDS W staging,
//   head-split store) -> flash attention with ALiBi (WMMA QK^T / PV, async
//   double-buffered K/V tiles, online softmax) -> output projection (f32).
// Async global->LDS copies use GLOBAL_LOAD_ASYNC_TO_LDS_B128 (ASYNCcnt) when
// the toolchain exposes the builtin; otherwise a synchronous 16B copy.
// ---------------------------------------------------------------------------

typedef _Float16 v16h __attribute__((ext_vector_type(16)));
typedef _Float16 v8h  __attribute__((ext_vector_type(8)));
typedef float    v8f  __attribute__((ext_vector_type(8)));

#define WMMA_F32_F16(a, b, c) \
  __builtin_amdgcn_wmma_f32_16x16x32_f16(false, (a), false, (b), (short)0, (c), false, false)

#if defined(__AMDGCN__) && __has_builtin(__builtin_amdgcn_global_load_async_to_lds_b128)
#define HAVE_ASYNC_LDS 1
#else
#define HAVE_ASYNC_LDS 0
#endif

#if HAVE_ASYNC_LDS
// Builtin prototype (from clang diagnostic): params are pointers to 16-byte
// int vectors in explicit address spaces: (v4i global*, v4i shared*, Ii, Ii).
typedef int v4i __attribute__((vector_size(16)));
typedef __attribute__((address_space(1))) v4i* gptr_b128;
typedef __attribute__((address_space(3))) v4i* lptr_b128;
#endif

// 16-byte global -> LDS copy (async when available; per-lane addresses).
__device__ __forceinline__ void cp16_to_lds(_Float16* lds_dst, const _Float16* gsrc) {
#if HAVE_ASYNC_LDS
  __builtin_amdgcn_global_load_async_to_lds_b128(
      (gptr_b128)gsrc, (lptr_b128)lds_dst, 0, 0);
#else
  *(v8h*)lds_dst = *(const v8h*)gsrc;
#endif
}

// Wait for this wave's outstanding async global->LDS copies.
__device__ __forceinline__ void wait_async_copies() {
#if HAVE_ASYNC_LDS
#if __has_builtin(__builtin_amdgcn_s_wait_asynccnt)
  __builtin_amdgcn_s_wait_asynccnt(0);
#else
  asm volatile("s_wait_asynccnt 0x0" ::: "memory");
#endif
#endif
}

constexpr int Bc  = 2;
constexpr int Sc  = 2048;
constexpr int Dc  = 1024;
constexpr int Hc  = 16;
constexpr int HDc = 64;
constexpr int Mrows = Bc * Sc;   // 4096 rows of x / attention output

// ---------------------------------------------------------------------------
// Elementwise fp32 -> fp16 conversion (grid-stride).
// ---------------------------------------------------------------------------
__global__ void cvt_f32_f16(const float* __restrict__ src,
                            _Float16* __restrict__ dst, int n) {
  int i = blockIdx.x * blockDim.x + threadIdx.x;
  int stride = gridDim.x * blockDim.x;
  for (; i < n; i += stride) dst[i] = (_Float16)src[i];
}

// ---------------------------------------------------------------------------
// C[m,n] = sum_k A[m,k] * W[n,k]   (y = x @ W^T), A:[4096,1024] W:[1024,1024]
// Block tile 128(M) x 64(N); per K=32 step the 64x32-half W tile is staged in
// LDS (async, double buffered) and shared by all 8 waves; A fragments load
// directly from global. 4 WMMAs per wave per K-step.
// HEADS=1: f16 store into [B,H,S,64].  HEADS=0: f32 store into [M,N].
// ---------------------------------------------------------------------------
template <int HEADS>
__global__ __launch_bounds__(256) void gemm_xwT(const _Float16* __restrict__ A,
                                                const _Float16* __restrict__ W,
                                                void* __restrict__ outv) {
  constexpr int K = Dc;
  constexpr int N = Dc;
  __shared__ _Float16 Wt[2][64][36];   // pad 36 halves: spread LDS banks

  const int wave = threadIdx.x >> 5;
  const int lane = threadIdx.x & 31;
  const int lh = lane & 15;        // lane-within-half: row (A) / col (B,C)
  const int hi = lane >> 4;        // lane half: selects K sub-range / M+8
  const int m0 = blockIdx.x * 128 + wave * 16;
  const int n0 = blockIdx.y * 64;

  // Staging split: 256 threads x 16B = one 64x32-half tile per step.
  const int tr = threadIdx.x >> 2;        // 0..63: W row within tile
  const int tc = (threadIdx.x & 3) * 8;   // 0,8,16,24: K offset (halves)
  const _Float16* wsrc = W + (size_t)(n0 + tr) * K + tc;

  cp16_to_lds(&Wt[0][tr][tc], wsrc);      // prologue: K-step 0

  const _Float16* arow = A + (size_t)(m0 + lh) * K;
  v8f acc[4] = {};

  int buf = 0;
  for (int k0 = 0; k0 < K; k0 += 32, buf ^= 1) {
    wait_async_copies();
    __syncthreads();                       // tile[buf] ready; tile[buf^1] free
    if (k0 + 32 < K) cp16_to_lds(&Wt[buf ^ 1][tr][tc], wsrc + k0 + 32);

    // A fragment 16x32: lanes 0-15 hold K 0..15; lanes 16-31 hold K 16..31
    v16h a;
    const int ka = k0 + hi * 16;
#pragma unroll
    for (int i = 0; i < 8; i++) {
      a[i]     = arow[ka + i];
      a[i + 8] = arow[ka + 8 + i];
    }
    // B fragments 32x16 from LDS: lane = N col; half selects K 0-15 / 16-31
#pragma unroll
    for (int j = 0; j < 4; j++) {
      const _Float16* wr = &Wt[buf][16 * j + lh][hi * 16];
      v16h b;
#pragma unroll
      for (int i = 0; i < 16; i++) b[i] = wr[i];
      acc[j] = WMMA_F32_F16(a, b, acc[j]);
    }
  }

  // C fragment: vgpr r, lane -> M = m0 + r + hi*8, N = n0 + 16j + lh
#pragma unroll
  for (int j = 0; j < 4; j++) {
#pragma unroll
    for (int r = 0; r < 8; r++) {
      const int m = m0 + r + hi * 8;
      const int n = n0 + 16 * j + lh;
      const float v = acc[j][r];
      if (HEADS) {
        const int b  = m >> 11;          // m = b*S + s, S=2048
        const int s  = m & (Sc - 1);
        const int h  = n >> 6;           // n = h*64 + hd
        const int hd = n & (HDc - 1);
        ((_Float16*)outv)[(((size_t)(b * Hc + h)) * Sc + s) * HDc + hd] = (_Float16)v;
      } else {
        ((float*)outv)[(size_t)m * N + n] = v;
      }
    }
  }
}

// ---------------------------------------------------------------------------
// Flash attention with ALiBi bias + causal mask.
// Block = 256 thr (8 waves) owns one (b,h) and a 128-query slab; each wave
// owns 16 queries. 32-key K/V tiles are async double-buffered through LDS.
// Row statistics live per lane-half (C-fragment row = vgpr r + 8*half).
// ---------------------------------------------------------------------------
__global__ __launch_bounds__(256) void flash_attn_alibi(
    const _Float16* __restrict__ Qh, const _Float16* __restrict__ Kh,
    const _Float16* __restrict__ Vh, _Float16* __restrict__ Oh) {
  __shared__ _Float16 Kt[2][32][68];    // pad 68 halves vs 64: bank spread
  __shared__ _Float16 Vt[2][32][68];
  __shared__ _Float16 Pt[8][16][32];    // per-wave P staging (C-frag -> A-frag)

  const int bh = blockIdx.y;
  const int h  = bh & (Hc - 1);
  const int b  = bh >> 4;
  const int wave = threadIdx.x >> 5;
  const int lane = threadIdx.x & 31;
  const int lh = lane & 15;
  const int hi = lane >> 4;
  const int q0 = blockIdx.x * 128 + wave * 16;

  const _Float16* Q  = Qh + (size_t)bh * Sc * HDc;
  const _Float16* Kp = Kh + (size_t)bh * Sc * HDc;
  const _Float16* Vp = Vh + (size_t)bh * Sc * HDc;

  const float slope = exp2f(-8.0f * (float)(h + 1) / (float)Hc);  // (2^8)^(-(h+1)/H)
  const float isq = 0.125f;                                       // 1/sqrt(64)

  // Staging split: 256 threads, K tile + V tile are 32x64 halves each.
  const int srow = threadIdx.x >> 3;        // 0..31
  const int sc8  = (threadIdx.x & 7) * 8;   // 0..56 step 8

  // Q A-fragments for hd chunks [0,32) and [32,64)
  v16h qa[2];
  {
    const _Float16* qrow = Q + (size_t)(q0 + lh) * HDc;
#pragma unroll
    for (int c = 0; c < 2; c++) {
      const int ka = 32 * c + hi * 16;
#pragma unroll
      for (int i = 0; i < 8; i++) {
        qa[c][i]     = qrow[ka + i];
        qa[c][i + 8] = qrow[ka + 8 + i];
      }
    }
  }

  v8f o[4] = {};
  float mrow[8], lrow[8];
#pragma unroll
  for (int r = 0; r < 8; r++) { mrow[r] = -__builtin_inff(); lrow[r] = 0.0f; }

  const int kkend = blockIdx.x * 128 + 127;  // last query index in block

  // Prologue: stage tile kk=0 into buffer 0.
  cp16_to_lds(&Kt[0][srow][sc8], Kp + (size_t)srow * HDc + sc8);
  cp16_to_lds(&Vt[0][srow][sc8], Vp + (size_t)srow * HDc + sc8);

  int buf = 0;
  for (int kk = 0; kk <= kkend; kk += 32, buf ^= 1) {
    wait_async_copies();
    __syncthreads();                     // tile[buf] ready; tile[buf^1] free
    if (kk + 32 <= kkend) {
      cp16_to_lds(&Kt[buf ^ 1][srow][sc8], Kp + (size_t)(kk + 32 + srow) * HDc + sc8);
      cp16_to_lds(&Vt[buf ^ 1][srow][sc8], Vp + (size_t)(kk + 32 + srow) * HDc + sc8);
    }

    if (kk <= q0 + 15) {  // wave has at least one unmasked key in this tile
      float p[2][8];
      // ---- scores: S = Q @ K^T (two 16-key N-tiles, K=32 over hd per WMMA)
#pragma unroll
      for (int j = 0; j < 2; j++) {
        v8f s = {};
#pragma unroll
        for (int c = 0; c < 2; c++) {
          v16h bf;  // B[k=hd][n=key]: lane=key, half selects hd 0-15/16-31 of chunk
#pragma unroll
          for (int i = 0; i < 16; i++) bf[i] = Kt[buf][16 * j + lh][32 * c + hi * 16 + i];
          s = WMMA_F32_F16(qa[c], bf, s);
        }
        // scale + ALiBi bias + causal mask, in C-fragment layout
#pragma unroll
        for (int r = 0; r < 8; r++) {
          const int q  = q0 + r + hi * 8;
          const int kx = kk + 16 * j + lh;
          const int d  = kx - q;
          const float v = s[r] * isq + slope * (d < 0 ? (float)d : 0.0f);
          p[j][r] = (d > 0) ? -__builtin_inff() : v;
        }
      }
      // ---- online softmax: butterfly reductions within 16-lane halves
#pragma unroll
      for (int r = 0; r < 8; r++) {
        float mr = fmaxf(p[0][r], p[1][r]);
#pragma unroll
        for (int off = 8; off >= 1; off >>= 1) mr = fmaxf(mr, __shfl_xor(mr, off, 32));
        const float mnew  = fmaxf(mrow[r], mr);
        const float alpha = __expf(mrow[r] - mnew);
        const float s0 = __expf(p[0][r] - mnew);
        const float s1 = __expf(p[1][r] - mnew);
        float rs = s0 + s1;
#pragma unroll
        for (int off = 8; off >= 1; off >>= 1) rs += __shfl_xor(rs, off, 32);
        lrow[r] = lrow[r] * alpha + rs;
        mrow[r] = mnew;
#pragma unroll
        for (int t = 0; t < 4; t++) o[t][r] *= alpha;
        p[0][r] = s0;
        p[1][r] = s1;
      }
      // ---- stage P through LDS to re-layout C-fragment -> A-fragment
#pragma unroll
      for (int j = 0; j < 2; j++)
#pragma unroll
        for (int r = 0; r < 8; r++)
          Pt[wave][r + hi * 8][16 * j + lh] = (_Float16)p[j][r];
      // same-wave LDS ops are in-order: no barrier needed for private region
      v16h pa;
      {
        const int ko = hi * 16;
#pragma unroll
        for (int i = 0; i < 8; i++) {
          pa[i]     = Pt[wave][lh][ko + i];
          pa[i + 8] = Pt[wave][lh][8 + ko + i];
        }
      }
      // ---- PV: O += P(16x32 keys) @ V(32 keys x 64 hd), 4 N-tiles
#pragma unroll
      for (int t = 0; t < 4; t++) {
        v16h vb;  // B[k=key][n=hd]: lane=hd col, half selects keys 0-15/16-31
#pragma unroll
        for (int i = 0; i < 16; i++) vb[i] = Vt[buf][hi * 16 + i][16 * t + lh];
        o[t] = WMMA_F32_F16(pa, vb, o[t]);
      }
    }
  }

  // Epilogue: normalize and store f16 into [B,S,D] (heads re-merged)
#pragma unroll
  for (int t = 0; t < 4; t++) {
#pragma unroll
    for (int r = 0; r < 8; r++) {
      const int q   = q0 + r + hi * 8;
      const int col = 16 * t + lh;
      const float val = o[t][r] / lrow[r];
      Oh[((size_t)b * Sc + q) * Dc + h * HDc + col] = (_Float16)val;
    }
  }
}

// ---------------------------------------------------------------------------
// Host-side launch. Workspace layout (halves):
//   xh[4M] | Wq[1M] Wk[1M] Wv[1M] Wo[1M] | Q[4M] K[4M] V[4M] | attn[4M]
// = 24M halves = 48 MB.
// ---------------------------------------------------------------------------
extern "C" void kernel_launch(void* const* d_in, const int* in_sizes, int n_in,
                              void* d_out, int out_size, void* d_ws, size_t ws_size,
                              hipStream_t stream) {
  (void)in_sizes; (void)n_in; (void)out_size; (void)ws_size;
  const float* x  = (const float*)d_in[0];
  const float* Wq = (const float*)d_in[1];
  const float* Wk = (const float*)d_in[2];
  const float* Wv = (const float*)d_in[3];
  const float* Wo = (const float*)d_in[4];
  float* out = (float*)d_out;

  const int XE = Mrows * Dc;  // 4,194,304
  const int WE = Dc * Dc;     // 1,048,576

  _Float16* xh  = (_Float16*)d_ws;
  _Float16* wqh = xh + (size_t)XE;
  _Float16* wkh = wqh + (size_t)WE;
  _Float16* wvh = wkh + (size_t)WE;
  _Float16* woh = wvh + (size_t)WE;
  _Float16* Qd  = woh + (size_t)WE;
  _Float16* Kd  = Qd + (size_t)XE;
  _Float16* Vd  = Kd + (size_t)XE;
  _Float16* Ad  = Vd + (size_t)XE;

  cvt_f32_f16<<<2048, 256, 0, stream>>>(x,  xh,  XE);
  cvt_f32_f16<<<1024, 256, 0, stream>>>(Wq, wqh, WE);
  cvt_f32_f16<<<1024, 256, 0, stream>>>(Wk, wkh, WE);
  cvt_f32_f16<<<1024, 256, 0, stream>>>(Wv, wvh, WE);
  cvt_f32_f16<<<1024, 256, 0, stream>>>(Wo, woh, WE);

  dim3 gg(Mrows / 128, Dc / 64);  // 32 x 16
  gemm_xwT<1><<<gg, 256, 0, stream>>>(xh, wqh, (void*)Qd);
  gemm_xwT<1><<<gg, 256, 0, stream>>>(xh, wkh, (void*)Kd);
  gemm_xwT<1><<<gg, 256, 0, stream>>>(xh, wvh, (void*)Vd);

  flash_attn_alibi<<<dim3(Sc / 128, Bc * Hc), 256, 0, stream>>>(Qd, Kd, Vd, Ad);

  gemm_xwT<0><<<gg, 256, 0, stream>>>(Ad, woh, (void*)out);
}
